// Encoder_72799695667611
// MI455X (gfx1250) — matrix operand order repaired and checked by reading, compiled, not run on hardware
//
#include <hip/hip_runtime.h>
#include <hip/hip_bf16.h>

typedef __attribute__((ext_vector_type(16))) __bf16 v16bf;
typedef __attribute__((ext_vector_type(8)))  float  v8f;
typedef __attribute__((ext_vector_type(4)))  unsigned uv4;

union Frag { v16bf v; unsigned u[8]; uv4 q[2]; };

#define EMBED 512
#define FDIM  2048
#define SEQ   2048
#define NBATCH 4
#define NHEAD 8
#define HDIM  64
#define MTOK  (NBATCH*SEQ)   // 8192
#define LOG2E 1.44269504088896340736f

// ---- 16-lane-row reductions via DPP16 (no LDS traffic) ----
__device__ __forceinline__ float row16_max(float x) {
    int t;
    t = __builtin_amdgcn_mov_dpp(__float_as_int(x), 0xB1, 0xF, 0xF, true);
    x = fmaxf(x, __int_as_float(t));
    t = __builtin_amdgcn_mov_dpp(__float_as_int(x), 0x4E, 0xF, 0xF, true);
    x = fmaxf(x, __int_as_float(t));
    t = __builtin_amdgcn_mov_dpp(__float_as_int(x), 0x124, 0xF, 0xF, true);
    x = fmaxf(x, __int_as_float(t));
    t = __builtin_amdgcn_mov_dpp(__float_as_int(x), 0x128, 0xF, 0xF, true);
    x = fmaxf(x, __int_as_float(t));
    return x;
}
__device__ __forceinline__ float row16_sum(float x) {
    int t;
    t = __builtin_amdgcn_mov_dpp(__float_as_int(x), 0xB1, 0xF, 0xF, true);
    x += __int_as_float(t);
    t = __builtin_amdgcn_mov_dpp(__float_as_int(x), 0x4E, 0xF, 0xF, true);
    x += __int_as_float(t);
    t = __builtin_amdgcn_mov_dpp(__float_as_int(x), 0x124, 0xF, 0xF, true);
    x += __int_as_float(t);
    t = __builtin_amdgcn_mov_dpp(__float_as_int(x), 0x128, 0xF, 0xF, true);
    x += __int_as_float(t);
    return x;
}

// ---------------- weight fp32 [K][N] -> bf16 transposed [N][K] ----------------
__global__ __launch_bounds__(256) void wt_kernel(const float* __restrict__ w,
                                                 __bf16* __restrict__ wT,
                                                 int K, int N) {
    int idx = blockIdx.x * 256 + threadIdx.x;
    int n = idx / K;
    int k = idx - n * K;
    wT[(size_t)n * K + k] = (__bf16)w[(size_t)k * N + n];
}

// ---------------- layernorm: fp32 row[512] -> bf16 row[512] ----------------
__global__ __launch_bounds__(128) void ln_kernel(const float* __restrict__ x,
                                                 const float* __restrict__ gg,
                                                 const float* __restrict__ bb,
                                                 __bf16* __restrict__ out) {
    __shared__ float red[8];
    const int row = blockIdx.x;
    const int t = threadIdx.x;
    const float* xr = x + (size_t)row * EMBED;
    float4 v = *(const float4*)&xr[t * 4];
    float s  = v.x + v.y + v.z + v.w;
    float s2 = v.x*v.x + v.y*v.y + v.z*v.z + v.w*v.w;
#pragma unroll
    for (int m = 16; m >= 1; m >>= 1) {
        s  += __shfl_xor(s,  m, 32);
        s2 += __shfl_xor(s2, m, 32);
    }
    if ((t & 31) == 0) { red[t >> 5] = s; red[4 + (t >> 5)] = s2; }
    __syncthreads();
    float ts  = red[0] + red[1] + red[2] + red[3];
    float ts2 = red[4] + red[5] + red[6] + red[7];
    float mu  = ts * (1.0f / EMBED);
    float var = ts2 * (1.0f / EMBED) - mu * mu;
    float r   = rsqrtf(var + 1e-5f);
    __bf16* o = out + (size_t)row * EMBED;
    const float* vp = (const float*)&v;
#pragma unroll
    for (int i = 0; i < 4; ++i) {
        int c = t * 4 + i;
        o[c] = (__bf16)((vp[i] - mu) * r * gg[c] + bb[c]);
    }
}

// ---------------- tiled GEMM (double-buffered LDS): C = A[M,K] * BT[N,K] + bias ----------------
// MODE 0: out bf16 -> [b,h,s,d] scaled      (Q: scale=log2e, K: scale=1)
// MODE 1: out bf16 -> [b,h,d,s]             (V projection, transposed)
// MODE 2: out bf16 -> spike (>= 2.0 ? 1:0)  (FFN1 + LIF threshold)
// MODE 3: out f32  -> acc + bias + resid    (FFN2 + residual)
template <int MODE>
__global__ __launch_bounds__(256) void gemm_kernel(const __bf16* __restrict__ A,
                                                   const __bf16* __restrict__ BT,
                                                   const float* __restrict__ bias,
                                                   __bf16* __restrict__ outb,
                                                   float* __restrict__ outf,
                                                   const float* __restrict__ resid,
                                                   float scale,
                                                   int M, int N, int K) {
    __shared__ __align__(16) __bf16 lA[2][128 * 40];
    __shared__ __align__(16) __bf16 lB[2][128 * 40];

    const int m0   = blockIdx.x * 128;
    const int n0   = blockIdx.y * 128;
    const int tid  = threadIdx.x;
    const int wave = tid >> 5;
    const int lane = tid & 31;
    const int lm   = lane & 15;
    const int g    = lane >> 4;
    const int wm   = wave >> 2;   // 0..1 : 64-row strip
    const int wn   = wave & 3;    // 0..3 : 32-col strip

    // per-thread staging coordinates (2 x 16B per matrix per tile)
    int rowi[2], qdi[2];
#pragma unroll
    for (int r = 0; r < 2; ++r) {
        int i2  = tid + r * 256;
        rowi[r] = i2 >> 2;
        qdi[r]  = i2 & 3;
    }

    const v8f vz = {0.f,0.f,0.f,0.f,0.f,0.f,0.f,0.f};
    v8f acc[4][2];
#pragma unroll
    for (int i = 0; i < 4; ++i)
#pragma unroll
        for (int j = 0; j < 2; ++j) acc[i][j] = vz;

    const int kTiles = K >> 5;

    // prologue: stage tile 0 into buffer 0
    uv4 ra[2], rb[2];
#pragma unroll
    for (int r = 0; r < 2; ++r) {
        ra[r] = *(const uv4*)&A [(size_t)(m0 + rowi[r]) * K + qdi[r] * 8];
        rb[r] = *(const uv4*)&BT[(size_t)(n0 + rowi[r]) * K + qdi[r] * 8];
    }
#pragma unroll
    for (int r = 0; r < 2; ++r) {
        *(uv4*)&lA[0][rowi[r] * 40 + qdi[r] * 8] = ra[r];
        *(uv4*)&lB[0][rowi[r] * 40 + qdi[r] * 8] = rb[r];
    }

    for (int kt = 0; kt < kTiles; ++kt) {
        const int cur = kt & 1;
        // issue next tile's global loads before the barrier (overlap with compute)
        if (kt + 1 < kTiles) {
#pragma unroll
            for (int r = 0; r < 2; ++r) {
                const __bf16* pa = &A [(size_t)(m0 + rowi[r]) * K + (kt + 1) * 32 + qdi[r] * 8];
                const __bf16* pb = &BT[(size_t)(n0 + rowi[r]) * K + (kt + 1) * 32 + qdi[r] * 8];
                ra[r] = *(const uv4*)pa;
                rb[r] = *(const uv4*)pb;
                __builtin_prefetch(pa + 64, 0, 3);   // 2 tiles ahead
                __builtin_prefetch(pb + 64, 0, 3);
            }
        }
        __syncthreads();    // buffer `cur` fully written

        Frag aF[4], bF[2];
#pragma unroll
        for (int i = 0; i < 4; ++i) {
            int rm = (wm * 64 + i * 16 + lm) * 40;
            aF[i].q[0] = *(const uv4*)&lA[cur][rm + g * 8];
            aF[i].q[1] = *(const uv4*)&lA[cur][rm + 16 + g * 8];
        }
#pragma unroll
        for (int jn = 0; jn < 2; ++jn) {
            int rn = (wn * 32 + jn * 16 + lm) * 40;
            bF[jn].q[0] = *(const uv4*)&lB[cur][rn + g * 8];
            bF[jn].q[1] = *(const uv4*)&lB[cur][rn + 16 + g * 8];
        }
#pragma unroll
        for (int i = 0; i < 4; ++i)
#pragma unroll
            for (int jn = 0; jn < 2; ++jn)
                acc[i][jn] = __builtin_amdgcn_wmma_f32_16x16x32_bf16(
                    false, aF[i].v, false, bF[jn].v, (short)0, acc[i][jn], false, false);

        // stage next tile into the alternate buffer (read next iter after barrier)
        if (kt + 1 < kTiles) {
#pragma unroll
            for (int r = 0; r < 2; ++r) {
                *(uv4*)&lA[cur ^ 1][rowi[r] * 40 + qdi[r] * 8] = ra[r];
                *(uv4*)&lB[cur ^ 1][rowi[r] * 40 + qdi[r] * 8] = rb[r];
            }
        }
    }

    // epilogue (C layout: n = lm, m = j + 8*g within each 16x16 tile)
#pragma unroll
    for (int i = 0; i < 4; ++i) {
        int mb = m0 + wm * 64 + i * 16;
#pragma unroll
        for (int jn = 0; jn < 2; ++jn) {
            int n = n0 + wn * 32 + jn * 16 + lm;
            float bvs = bias[n];
#pragma unroll
            for (int j = 0; j < 8; ++j) {
                int m = mb + j + 8 * g;
                float val = acc[i][jn][j] + bvs;
                if (MODE == 0) {
                    int b = m >> 11, s = m & (SEQ - 1), h = n >> 6, dd = n & 63;
                    outb[(((size_t)(b * NHEAD + h) * SEQ + s) << 6) + dd] = (__bf16)(val * scale);
                } else if (MODE == 1) {
                    int b = m >> 11, s = m & (SEQ - 1), h = n >> 6, dd = n & 63;
                    outb[((size_t)(b * NHEAD + h) * HDIM + dd) * SEQ + s] = (__bf16)val;
                } else if (MODE == 2) {
                    outb[(size_t)m * N + n] = (val >= 2.0f) ? (__bf16)1.0f : (__bf16)0.0f;
                } else {
                    size_t o = (size_t)m * N + n;
                    outf[o] = val + resid[o];
                }
            }
        }
    }
}

// ---------------- flash attention: one wave per TWO 16-query tiles ----------------
// Q is pre-scaled by log2e, softmax runs in exp2 domain (shift-invariant).
// Per-u processing is sequential to keep VGPR count under 256 (no MSB mode).
__global__ __launch_bounds__(128, 4) void attn_kernel(const __bf16* __restrict__ Q,
                                                      const __bf16* __restrict__ Kh,
                                                      const __bf16* __restrict__ VT,
                                                      const float* __restrict__ x,
                                                      float* __restrict__ x1) {
    __shared__ __align__(16) __bf16 pbuf[4][2][16 * 32];
    const int wid  = threadIdx.x >> 5;
    const int lane = threadIdx.x & 31;
    const int lm   = lane & 15;
    const int g    = lane >> 4;
    const int gw   = blockIdx.x * 4 + wid;   // 0..2047
    const int bh   = gw >> 6;                // batch*head 0..31
    const int qp   = gw & 63;                // pair of query tiles, q0 = qp*32
    const size_t headOff = (size_t)bh * SEQ * HDIM;
    const __bf16* vbase = VT + (size_t)bh * HDIM * SEQ;

    // Q fragments for both query tiles (A-layout, two contiguous 8-half runs)
    Frag aQ[2][2];
#pragma unroll
    for (int u = 0; u < 2; ++u) {
        const __bf16* qrow = Q + headOff + (size_t)(qp * 32 + u * 16 + lm) * HDIM;
        aQ[u][0].q[0] = *(const uv4*)&qrow[g * 8];
        aQ[u][0].q[1] = *(const uv4*)&qrow[16 + g * 8];
        aQ[u][1].q[0] = *(const uv4*)&qrow[32 + g * 8];
        aQ[u][1].q[1] = *(const uv4*)&qrow[48 + g * 8];
    }

    const v8f vz = {0.f,0.f,0.f,0.f,0.f,0.f,0.f,0.f};
    v8f oacc[2][4] = {{vz, vz, vz, vz}, {vz, vz, vz, vz}};
    float mx[16], ls[16];
#pragma unroll
    for (int j = 0; j < 16; ++j) { mx[j] = -3.0e38f; ls[j] = 0.0f; }

    for (int kc = 0; kc < SEQ / 32; ++kc) {
        const int k0 = kc * 32;
        // ---- K fragments for this 32-key chunk (shared by both query tiles) ----
        const __bf16* kr0 = Kh + headOff + (size_t)(k0 + lm) * HDIM;
        const __bf16* kr1 = kr0 + 16 * HDIM;
        Frag b0a, b0b, b1a, b1b;
        b0a.q[0] = *(const uv4*)&kr0[g * 8];      b0a.q[1] = *(const uv4*)&kr0[16 + g * 8];
        b0b.q[0] = *(const uv4*)&kr0[32 + g * 8]; b0b.q[1] = *(const uv4*)&kr0[48 + g * 8];
        b1a.q[0] = *(const uv4*)&kr1[g * 8];      b1a.q[1] = *(const uv4*)&kr1[16 + g * 8];
        b1b.q[0] = *(const uv4*)&kr1[32 + g * 8]; b1b.q[1] = *(const uv4*)&kr1[48 + g * 8];
        if (kc + 1 < SEQ / 32) {
            __builtin_prefetch(kr0 + 32 * HDIM, 0, 3);
            __builtin_prefetch(kr1 + 32 * HDIM, 0, 3);
        }
        // ---- per query tile: scores + online softmax (keeps one s-pair live) ----
#pragma unroll
        for (int u = 0; u < 2; ++u) {
            v8f s0, s1;
            s0 = __builtin_amdgcn_wmma_f32_16x16x32_bf16(false, aQ[u][0].v, false, b0a.v, (short)0, vz, false, false);
            s0 = __builtin_amdgcn_wmma_f32_16x16x32_bf16(false, aQ[u][1].v, false, b0b.v, (short)0, s0, false, false);
            s1 = __builtin_amdgcn_wmma_f32_16x16x32_bf16(false, aQ[u][0].v, false, b1a.v, (short)0, vz, false, false);
            s1 = __builtin_amdgcn_wmma_f32_16x16x32_bf16(false, aQ[u][1].v, false, b1b.v, (short)0, s1, false, false);
            __bf16* pb = pbuf[wid][u];
#pragma unroll
            for (int j = 0; j < 8; ++j) {
                float a = s0[j], b = s1[j];
                float t  = row16_max(fmaxf(a, b));
                int   sj = u * 8 + j;
                float nm = fmaxf(mx[sj], t);
                float sc = __builtin_amdgcn_exp2f(mx[sj] - nm);
                float p0 = __builtin_amdgcn_exp2f(a - nm);
                float p1 = __builtin_amdgcn_exp2f(b - nm);
                float rs = row16_sum(p0 + p1);
                ls[sj] = ls[sj] * sc + rs;
                mx[sj] = nm;
                oacc[u][0][j] *= sc; oacc[u][1][j] *= sc;
                oacc[u][2][j] *= sc; oacc[u][3][j] *= sc;
                int mrow = j + 8 * g;
                pb[mrow * 32 + lm]      = (__bf16)p0;
                pb[mrow * 32 + 16 + lm] = (__bf16)p1;
            }
        }
        // ---- reload P tiles in A-layout: two b128 reads per lane each ----
        Frag aP[2];
#pragma unroll
        for (int u = 0; u < 2; ++u) {
            aP[u].q[0] = *(const uv4*)&pbuf[wid][u][lm * 32 + g * 8];
            aP[u].q[1] = *(const uv4*)&pbuf[wid][u][lm * 32 + 16 + g * 8];
        }
        // ---- context: V fragments shared by both query tiles ----
#pragma unroll
        for (int dc = 0; dc < 4; ++dc) {
            const __bf16* vr = vbase + (size_t)(dc * 16 + lm) * SEQ + k0;
            Frag bV;
            bV.q[0] = *(const uv4*)&vr[g * 8];
            bV.q[1] = *(const uv4*)&vr[16 + g * 8];
            if (kc + 1 < SEQ / 32) __builtin_prefetch(vr + 32, 0, 3);
#pragma unroll
            for (int u = 0; u < 2; ++u)
                oacc[u][dc] = __builtin_amdgcn_wmma_f32_16x16x32_bf16(
                    false, aP[u].v, false, bV.v, (short)0, oacc[u][dc], false, false);
        }
    }

    // ---- normalize + residual to x1 (fp32) ----
    const int b = bh >> 3, h = bh & 7;
#pragma unroll
    for (int u = 0; u < 2; ++u) {
#pragma unroll
        for (int j = 0; j < 8; ++j) {
            float inv = 1.0f / ls[u * 8 + j];
            int tok = b * SEQ + qp * 32 + u * 16 + j + 8 * g;
#pragma unroll
            for (int dc = 0; dc < 4; ++dc) {
                int feat = h * HDIM + dc * 16 + lm;
                size_t off = (size_t)tok * EMBED + feat;
                x1[off] = x[off] + oacc[u][dc][j] * inv;
            }
        }
    }
}

// ---------------- host orchestration ----------------
extern "C" void kernel_launch(void* const* d_in, const int* in_sizes, int n_in,
                              void* d_out, int out_size, void* d_ws, size_t ws_size,
                              hipStream_t stream) {
    (void)in_sizes; (void)n_in; (void)out_size; (void)ws_size;
    const float* x   = (const float*)d_in[0];
    const float* wq  = (const float*)d_in[1];
    const float* bq  = (const float*)d_in[2];
    const float* wk  = (const float*)d_in[3];
    const float* bk  = (const float*)d_in[4];
    const float* wv  = (const float*)d_in[5];
    const float* bv  = (const float*)d_in[6];
    const float* w1  = (const float*)d_in[7];
    const float* b1  = (const float*)d_in[8];
    const float* w2  = (const float*)d_in[9];
    const float* b2  = (const float*)d_in[10];
    const float* g1  = (const float*)d_in[11];
    const float* be1 = (const float*)d_in[12];
    const float* g2  = (const float*)d_in[13];
    const float* be2 = (const float*)d_in[14];
    float* out = (float*)d_out;

    char* ws = (char*)d_ws;
    size_t off = 0;
    __bf16* wqT = (__bf16*)(ws + off); off += (size_t)EMBED * EMBED * 2;
    __bf16* wkT = (__bf16*)(ws + off); off += (size_t)EMBED * EMBED * 2;
    __bf16* wvT = (__bf16*)(ws + off); off += (size_t)EMBED * EMBED * 2;
    __bf16* w1T = (__bf16*)(ws + off); off += (size_t)EMBED * FDIM * 2;
    __bf16* w2T = (__bf16*)(ws + off); off += (size_t)FDIM * EMBED * 2;
    __bf16* n1  = (__bf16*)(ws + off); off += (size_t)MTOK * EMBED * 2;
    __bf16* qb  = (__bf16*)(ws + off); off += (size_t)MTOK * EMBED * 2;
    __bf16* kb  = (__bf16*)(ws + off); off += (size_t)MTOK * EMBED * 2;
    __bf16* vTb = (__bf16*)(ws + off); off += (size_t)MTOK * EMBED * 2;
    float*  x1  = (float*)(ws + off);  off += (size_t)MTOK * EMBED * 4;
    __bf16* n2  = (__bf16*)(ws + off); off += (size_t)MTOK * EMBED * 2;
    __bf16* spk = (__bf16*)(ws + off); off += (size_t)MTOK * FDIM * 2;

    // weights -> bf16 transposed
    wt_kernel<<<(EMBED * EMBED) / 256, 256, 0, stream>>>(wq, wqT, EMBED, EMBED);
    wt_kernel<<<(EMBED * EMBED) / 256, 256, 0, stream>>>(wk, wkT, EMBED, EMBED);
    wt_kernel<<<(EMBED * EMBED) / 256, 256, 0, stream>>>(wv, wvT, EMBED, EMBED);
    wt_kernel<<<(EMBED * FDIM)  / 256, 256, 0, stream>>>(w1, w1T, EMBED, FDIM);
    wt_kernel<<<(FDIM * EMBED)  / 256, 256, 0, stream>>>(w2, w2T, FDIM, EMBED);

    // pre-norm 1
    ln_kernel<<<MTOK, 128, 0, stream>>>(x, g1, be1, n1);

    // QKV projections (Q pre-scaled by log2e for exp2-domain softmax)
    dim3 gqkv(MTOK / 128, EMBED / 128);
    gemm_kernel<0><<<gqkv, 256, 0, stream>>>(n1, wqT, bq, qb,  nullptr, nullptr, LOG2E, MTOK, EMBED, EMBED);
    gemm_kernel<0><<<gqkv, 256, 0, stream>>>(n1, wkT, bk, kb,  nullptr, nullptr, 1.0f,  MTOK, EMBED, EMBED);
    gemm_kernel<1><<<gqkv, 256, 0, stream>>>(n1, wvT, bv, vTb, nullptr, nullptr, 1.0f,  MTOK, EMBED, EMBED);

    // attention + residual -> x1  (one wave per 32 queries)
    attn_kernel<<<(NBATCH * NHEAD * (SEQ / 32)) / 4, 128, 0, stream>>>(qb, kb, vTb, x, x1);

    // pre-norm 2
    ln_kernel<<<MTOK, 128, 0, stream>>>(x1, g2, be2, n2);

    // FFN1 + spike threshold
    gemm_kernel<2><<<dim3(MTOK / 128, FDIM / 128), 256, 0, stream>>>(
        n2, w1T, b1, spk, nullptr, nullptr, 1.0f, MTOK, FDIM, EMBED);

    // FFN2 + bias + residual -> out
    gemm_kernel<3><<<dim3(MTOK / 128, EMBED / 128), 256, 0, stream>>>(
        spk, w2T, b2, nullptr, out, x1, 1.0f, MTOK, EMBED, FDIM);
}
